// ActSpanDecoder_56435870269718
// MI455X (gfx1250) — matrix operand order repaired and checked by reading, not compiled, not run on hardware
//
#include <hip/hip_runtime.h>
#include <hip/hip_bf16.h>
#include <math.h>

// Problem constants
#define B_    128
#define TU    256
#define TB    128
#define TP    64
#define H_    512
#define E_    512
#define V_    3000
#define VOOV_ 3400
#define PTR_  32
#define NEG_  (-1e20f)
#define XDIM  (E_ + 3 * H_ + PTR_)   // 2080
#define GH3   (3 * H_)               // 1536
#define CT    8                      // enc rows per async tile

typedef __attribute__((ext_vector_type(16))) __bf16 v16bf;
typedef __attribute__((ext_vector_type(8)))  float  v8f;

// Async LDS-DMA path only if the toolchain exposes the gfx1250 builtins.
#if defined(__gfx1250__) && \
    __has_builtin(__builtin_amdgcn_global_load_async_to_lds_b128) && \
    __has_builtin(__builtin_amdgcn_s_wait_asynccnt)
#define HAVE_ASYNC_LDS 1
#else
#define HAVE_ASYNC_LDS 0
#endif

#if HAVE_ASYNC_LDS
// The builtin's parameter type (from the compiler diagnostic) is a pointer to
// a GCC-style 16-byte int vector, in AS1 (global) / AS3 (LDS).
typedef int v4i_b128 __attribute__((vector_size(4 * sizeof(int))));
typedef __attribute__((address_space(1))) v4i_b128* gas_ptr;
typedef __attribute__((address_space(3))) v4i_b128* las_ptr;
__device__ __forceinline__ void async_b128(const float* g, float* l) {
  __builtin_amdgcn_global_load_async_to_lds_b128((gas_ptr)(void*)g,
                                                 (las_ptr)(void*)l, 0, 0);
}
#endif

// ---- float -> bf16 (round to nearest even) ----
__device__ __forceinline__ __bf16 f2bf(float x) {
  unsigned u = __builtin_bit_cast(unsigned, x);
  unsigned r = u + 0x7FFFu + ((u >> 16) & 1u);
  unsigned short h = (unsigned short)(r >> 16);
  return __builtin_bit_cast(__bf16, h);
}

// ---- A-fragment gather from LDS, 16x32 bf16, per CDNA5 ISA layout ----
// lane m = lane%16 holds row m; half = lane/16 selects K subrange.
// element i -> k = 16*(i/8) + (i%8) + 8*half  (contiguous 8 -> ds_load_b128)
__device__ __forceinline__ v16bf load_a_frag(const __bf16* As, int ld, int kBase, int lane) {
  int m = lane & 15;
  int half = lane >> 4;
  v16bf a;
#pragma unroll
  for (int i = 0; i < 16; ++i) {
    int k = ((i >> 3) << 4) + (i & 7) + (half << 3);
    a[i] = As[m * ld + kBase + k];
  }
  return a;
}

__device__ __forceinline__ v8f wmma_bf16(v16bf a, v16bf b, v8f c) {
  return __builtin_amdgcn_wmma_f32_16x16x32_bf16(false, a, false, b, (short)0, c,
                                                 false, false);
}

// =====================================================================
// Pack weight W (Nrows x K, row-major f32, optional column offset) into
// bf16 B-fragment order: dst[((nt*KC + kc)*32 + lane)*16 + i] =
//   W[nt*16 + lane%16][kc*32 + 16*(i/8) + (i%8) + 8*(lane/16)]
// so each wave's B fragment is one contiguous 32-byte load per lane.
// =====================================================================
__global__ void pack_w_kernel(const float* __restrict__ src, int ldsrc, int colOff,
                              int Nrows, int Ntiles, int KC, __bf16* __restrict__ dst) {
  long long idx = (long long)blockIdx.x * 256 + threadIdx.x;   // one per (nt,kc,lane)
  long long total = (long long)Ntiles * KC * 32;
  if (idx >= total) return;
  int lane = (int)(idx & 31);
  long long t = idx >> 5;
  int kc = (int)(t % KC);
  int nt = (int)(t / KC);
  int row = nt * 16 + (lane & 15);
  if (row >= Nrows) row = Nrows - 1;                            // pad tiles: clamp (stores guarded later)
  int colBase = kc * 32 + ((lane >> 4) << 3);
  __bf16* d = dst + idx * 16;
  const float* s = src + (long long)row * ldsrc + colOff + colBase;
#pragma unroll
  for (int i = 0; i < 16; ++i) d[i] = f2bf(s[(i & 7) + ((i >> 3) << 4)]);
}

// =====================================================================
// Generic GEMM: C = A(f32 MxK) * Wpackᵀ + bias, C f32 (M x N, ldc)
// block = 256 thr (8 waves), 16 rows per block, 8 n-tiles per block.
// A staged f32->bf16 into LDS per 512-wide K panel.
// =====================================================================
__global__ void gemm_bf16_kernel(const float* __restrict__ A, int lda,
                                 const __bf16* __restrict__ Wp,
                                 const float* __restrict__ bias,
                                 float* __restrict__ C, int ldc,
                                 int M, int N, int K) {
  __shared__ __bf16 As[16][512];
  int lane = threadIdx.x & 31, wave = threadIdx.x >> 5;
  int mBase = blockIdx.x * 16;
  int nt = blockIdx.y * 8 + wave;
  int KC = (K + 31) >> 5;
  v8f acc = {};
  for (int kp = 0; kp < K; kp += 512) {
    int panel = min(512, K - kp);
    for (int idx = threadIdx.x; idx < 16 * panel; idx += 256) {
      int r = idx / panel, c = idx - r * panel;
      int gr = mBase + r;
      float v = (gr < M) ? A[(long long)gr * lda + kp + c] : 0.f;
      As[r][c] = f2bf(v);
    }
    __syncthreads();
    int chunks = panel >> 5;
    for (int kc = 0; kc < chunks; ++kc) {
      v16bf a = load_a_frag(&As[0][0], 512, kc * 32, lane);
      long long base = (((long long)nt * KC + ((kp >> 5) + kc)) * 32 + lane) * 16;
      __builtin_prefetch(Wp + base + 32 * 16, 0, 1);   // global_prefetch_b8 on next chunk
      v16bf bfrag = *(const v16bf*)(Wp + base);
      acc = wmma_bf16(a, bfrag, acc);
    }
    __syncthreads();
  }
  int n = nt * 16 + (lane & 15);
  int half = lane >> 4;
  float bv = (n < N) ? bias[n] : 0.f;
#pragma unroll
  for (int i = 0; i < 8; ++i) {
    int m = mBase + half * 8 + i;
    if (m < M && n < N) C[(long long)m * ldc + n] = acc[i] + bv;
  }
}

// =====================================================================
// Fused  out[row] = sum_n tanh( (A@Wᵀ)[row,n] + addP[b,n] ) * mulP[b,n]
//   attention scores: addP = hW1+b (stride H), mulP = v_w (stride 0)
//   copy scores     : addP = Wcopy_b (stride 0), mulP = hnew (stride H)
// rows = flattened (b,t); K = N = 512; never materializes the e tensor.
// =====================================================================
__global__ void fused_tanh_dot_kernel(const float* __restrict__ A,
                                      const __bf16* __restrict__ Wp,
                                      const float* __restrict__ addP, int addStride,
                                      const float* __restrict__ mulP, int mulStride,
                                      float* __restrict__ out, int T) {
  __shared__ __bf16 As[16][H_];
  __shared__ float rowsum[8][16];
  int lane = threadIdx.x & 31, wave = threadIdx.x >> 5;
  int rBase = blockIdx.x * 16;
  for (int idx = threadIdx.x; idx < 16 * H_; idx += 256) {
    int r = idx >> 9, c = idx & (H_ - 1);
    As[r][c] = f2bf(A[(long long)(rBase + r) * H_ + c]);
  }
  if (lane < 16) rowsum[wave][lane] = 0.f;
  __syncthreads();
  int half = lane >> 4;
  int b = rBase / T;  // T is a multiple of 16 -> all 16 rows share one batch index
#pragma unroll 1
  for (int nti = 0; nti < 4; ++nti) {
    int nt = wave + nti * 8;
    v8f acc = {};
#pragma unroll 1
    for (int kc = 0; kc < 16; ++kc) {
      v16bf a = load_a_frag(&As[0][0], H_, kc * 32, lane);
      long long base = (((long long)nt * 16 + kc) * 32 + lane) * 16;
      __builtin_prefetch(Wp + base + 32 * 16, 0, 1);
      v16bf bfrag = *(const v16bf*)(Wp + base);
      acc = wmma_bf16(a, bfrag, acc);
    }
    int n = nt * 16 + (lane & 15);
    float av = addP[(long long)b * addStride + n];
    float mv = mulP[(long long)b * mulStride + n];
#pragma unroll
    for (int i = 0; i < 8; ++i) {
      float v = tanhf(acc[i] + av) * mv;
#pragma unroll
      for (int off = 1; off < 16; off <<= 1) v += __shfl_xor(v, off);
      if ((lane & 15) == 0) rowsum[wave][i + half * 8] += v;  // lanes 0 & 16: distinct rows
    }
  }
  __syncthreads();
  if (threadIdx.x < 16) {
    float s = 0.f;
#pragma unroll
    for (int w = 0; w < 8; ++w) s += rowsum[w][threadIdx.x];
    out[rBase + threadIdx.x] = s;
  }
}

// =====================================================================
// Masked softmax over T then ctx[b,h] = sum_t a[t]*enc[b,t,h].
// One block per batch row. Context accumulation streams enc through LDS
// with double-buffered async-to-LDS DMA (gfx1250) when available.
// =====================================================================
__global__ void softmax_ctx_kernel(const float* __restrict__ score,
                                   const int* __restrict__ ids,
                                   const float* __restrict__ enc,
                                   float* __restrict__ ctx, int T) {
  __shared__ float aw[TU];
  __shared__ float red[8];
  __shared__ float bc;
  int b = blockIdx.x, tid = threadIdx.x;
  int lane = tid & 31, wave = tid >> 5;
  float s = -INFINITY;
  if (tid < T) s = (ids[b * T + tid] == 0) ? NEG_ : score[b * T + tid];
  float m = s;
  for (int off = 16; off > 0; off >>= 1) m = fmaxf(m, __shfl_xor(m, off));
  if (lane == 0) red[wave] = m;
  __syncthreads();
  if (tid == 0) {
    float mm = red[0];
    for (int w = 1; w < 8; ++w) mm = fmaxf(mm, red[w]);
    bc = mm;
  }
  __syncthreads();
  m = bc;
  float e = (tid < T) ? expf(s - m) : 0.f;
  float sum = e;
  for (int off = 16; off > 0; off >>= 1) sum += __shfl_xor(sum, off);
  if (lane == 0) red[wave] = sum;
  __syncthreads();
  if (tid == 0) {
    float ss = 0.f;
    for (int w = 0; w < 8; ++w) ss += red[w];
    bc = ss;
  }
  __syncthreads();
  float inv = 1.f / bc;
  if (tid < T) aw[tid] = e * inv;
  __syncthreads();

#if HAVE_ASYNC_LDS
  // Double-buffered async DMA: DMA tile it+1 while accumulating tile it.
  __shared__ float tile[2][CT][H_];
  float acc0 = 0.f, acc1 = 0.f;
  int ntile = T / CT;                      // T in {64,128,256} -> exact
  const float* src = enc + (long long)b * T * H_;
#pragma unroll
  for (int q = 0; q < 4; ++q) {            // tile 0: 4 async b128 per thread
    int idx = tid + q * 256;
    async_b128(src + idx * 4, &tile[0][0][0] + idx * 4);
  }
  for (int it = 0; it < ntile; ++it) {
    int cur = it & 1;
    if (it + 1 < ntile) {
      const float* s2 = src + (long long)(it + 1) * CT * H_;
      float* dst = &tile[cur ^ 1][0][0];
#pragma unroll
      for (int q = 0; q < 4; ++q) {
        int idx = tid + q * 256;
        async_b128(s2 + idx * 4, dst + idx * 4);
      }
      __builtin_amdgcn_s_wait_asynccnt(4);  // async loads complete in order
    } else {
      __builtin_amdgcn_s_wait_asynccnt(0);
    }
    __syncthreads();
#pragma unroll
    for (int t = 0; t < CT; ++t) {
      float w = aw[it * CT + t];
      acc0 += w * tile[cur][t][tid];
      acc1 += w * tile[cur][t][tid + 256];
    }
    __syncthreads();   // buffer 'cur' free for tile it+2's DMA
  }
  ctx[(long long)b * H_ + tid] = acc0;
  ctx[(long long)b * H_ + tid + 256] = acc1;
#else
  for (int h = tid; h < H_; h += 256) {
    float acc = 0.f;
    for (int t = 0; t < T; ++t)
      acc += aw[t] * enc[((long long)b * T + t) * H_ + h];
    ctx[(long long)b * H_ + h] = acc;
  }
#endif
}

// ---- assemble GRU input x = [emb | ctx_u | ctx_b | ctx_p | db] ----
__global__ void build_x_kernel(const int* __restrict__ w, const float* __restrict__ emb,
                               const float* __restrict__ cu, const float* __restrict__ cb,
                               const float* __restrict__ cp, const float* __restrict__ db,
                               float* __restrict__ x) {
  int b = blockIdx.x;
  for (int j = threadIdx.x; j < XDIM; j += 256) {
    float v;
    if (j < E_)              v = emb[(long long)w[b] * E_ + j];
    else if (j < E_ + H_)    v = cu[b * H_ + (j - E_)];
    else if (j < E_ + 2*H_)  v = cb[b * H_ + (j - E_ - H_)];
    else if (j < E_ + 3*H_)  v = cp[b * H_ + (j - E_ - 2*H_)];
    else                     v = db[b * PTR_ + (j - E_ - 3*H_)];
    x[(long long)b * XDIM + j] = v;
  }
}

// ---- GRU gate math ----
__global__ void gru_gates_kernel(const float* __restrict__ gi, const float* __restrict__ gh,
                                 const float* __restrict__ h0, float* __restrict__ hnew) {
  int idx = blockIdx.x * 256 + threadIdx.x;
  if (idx >= B_ * H_) return;
  int b = idx >> 9, h = idx & (H_ - 1);
  long long base = (long long)b * GH3;
  float ir = gi[base + h], iz = gi[base + H_ + h], in = gi[base + 2 * H_ + h];
  float hr = gh[base + h], hz = gh[base + H_ + h], hn = gh[base + 2 * H_ + h];
  float r = 1.f / (1.f + expf(-(ir + hr)));
  float z = 1.f / (1.f + expf(-(iz + hz)));
  float n = tanhf(in + r * hn);
  hnew[idx] = (1.f - z) * n + z * h0[idx];
}

// =====================================================================
// Finalize per batch row: rebuild cps from nounk scatter (deterministic,
// serialized), log-softmax over [gen | cps] (6128), exp-scatter-add into
// OOV slots, logaddexp for the vocab slots.
// =====================================================================
__global__ void finalize_kernel(const float* __restrict__ gen,
                                const float* __restrict__ cp_raw,
                                const int* __restrict__ bspn_ids,
                                const int* __restrict__ nounk,
                                float* __restrict__ out) {
  __shared__ float cps[V_ + TB];
  __shared__ float addv[VOOV_];
  __shared__ float red[8];
  __shared__ float bc;
  int b = blockIdx.x, tid = threadIdx.x;
  int lane = tid & 31, wave = tid >> 5;
  for (int i = tid; i < V_ + TB; i += 256) cps[i] = 0.f;
  for (int i = tid; i < VOOV_; i += 256) addv[i] = 0.f;
  __syncthreads();
  if (tid == 0) {
    for (int t = 0; t < TB; ++t) {
      float v = (bspn_ids[b * TB + t] == 0) ? NEG_ : cp_raw[b * TB + t];
      int nk = nounk[b * TB + t];
      int c = (nk < V_) ? nk : (V_ + t);
      cps[c] += v;
    }
  }
  __syncthreads();
  float m = -INFINITY;
  for (int i = tid; i < V_; i += 256) m = fmaxf(m, gen[(long long)b * V_ + i]);
  for (int i = tid; i < V_ + TB; i += 256) m = fmaxf(m, cps[i]);
  for (int off = 16; off > 0; off >>= 1) m = fmaxf(m, __shfl_xor(m, off));
  if (lane == 0) red[wave] = m;
  __syncthreads();
  if (tid == 0) {
    float mm = red[0];
    for (int w = 1; w < 8; ++w) mm = fmaxf(mm, red[w]);
    bc = mm;
  }
  __syncthreads();
  m = bc;
  float s = 0.f;
  for (int i = tid; i < V_; i += 256) s += expf(gen[(long long)b * V_ + i] - m);
  for (int i = tid; i < V_ + TB; i += 256) s += expf(cps[i] - m);
  for (int off = 16; off > 0; off >>= 1) s += __shfl_xor(s, off);
  if (lane == 0) red[wave] = s;
  __syncthreads();
  if (tid == 0) {
    float ss = 0.f;
    for (int w = 0; w < 8; ++w) ss += red[w];
    bc = m + logf(ss);
  }
  __syncthreads();
  float logZ = bc;
  if (tid == 0) {
    for (int t = 0; t < TB; ++t) {
      int nk = nounk[b * TB + t];
      if (nk >= V_) addv[nk] += expf(cps[V_ + t] - logZ);
    }
  }
  __syncthreads();
  for (int v = tid; v < VOOV_; v += 256) {
    float o;
    if (v < V_) {
      float a = gen[(long long)b * V_ + v] - logZ;
      float c = cps[v] - logZ;
      float mx = fmaxf(a, c), mn = fminf(a, c);
      o = mx + log1pf(expf(mn - mx));
    } else {
      float a = addv[v];
      o = (a > 0.f) ? logf(fmaxf(a, 1e-38f)) : NEG_;
    }
    out[(long long)b * VOOV_ + v] = o;
  }
}

// =====================================================================
extern "C" void kernel_launch(void* const* d_in, const int* in_sizes, int n_in,
                              void* d_out, int out_size, void* d_ws, size_t ws_size,
                              hipStream_t stream) {
  (void)in_sizes; (void)n_in; (void)out_size; (void)ws_size;
  const int*   dec_last_w  = (const int*)d_in[0];
  const float* dec_last_h  = (const float*)d_in[1];   // (1,B,H) -> (B,H)
  const float* usdx_h      = (const float*)d_in[2];
  const float* bspn_h      = (const float*)d_in[3];
  const float* pvaspn_h    = (const float*)d_in[4];
  const float* db          = (const float*)d_in[5];
  const int*   usdx_ids    = (const int*)d_in[6];
  const int*   bspn_ids    = (const int*)d_in[7];
  const int*   pvaspn_ids  = (const int*)d_in[8];
  const int*   bspn_nounk  = (const int*)d_in[9];
  /* d_in[10] bspn_onehot: reconstructed from nounk, never read */
  const float* emb_table   = (const float*)d_in[11];
  const float* attn_W      = (const float*)d_in[12];
  const float* attn_b      = (const float*)d_in[13];
  const float* v_w         = (const float*)d_in[14];
  const float* Wcopy_w     = (const float*)d_in[15];
  const float* Wcopy_b     = (const float*)d_in[16];
  const float* Wgen_w      = (const float*)d_in[17];
  const float* Wgen_b      = (const float*)d_in[18];
  const float* gru_W_ih    = (const float*)d_in[19];
  const float* gru_W_hh    = (const float*)d_in[20];
  const float* gru_b_ih    = (const float*)d_in[21];
  const float* gru_b_hh    = (const float*)d_in[22];
  float* out = (float*)d_out;

  char* ws = (char*)d_ws;
  size_t off = 0;
  auto alloc = [&](size_t bytes) {
    void* p = ws + off;
    off = (off + bytes + 255) & ~(size_t)255;
    return p;
  };
  const int KC_H = H_ / 32;    // 16
  const int KC_X = XDIM / 32;  // 65
  auto packBytes = [](int ntiles, int kc) {
    return (size_t)ntiles * kc * 32 * 16 * sizeof(__bf16);
  };

  __bf16* W1p   = (__bf16*)alloc(packBytes(32,  KC_H));
  __bf16* W2p   = (__bf16*)alloc(packBytes(32,  KC_H));
  __bf16* Wcpyp = (__bf16*)alloc(packBytes(32,  KC_H));
  __bf16* Wihp  = (__bf16*)alloc(packBytes(96,  KC_X));
  __bf16* Whhp  = (__bf16*)alloc(packBytes(96,  KC_H));
  __bf16* Wgenp = (__bf16*)alloc(packBytes(192, KC_H));
  float* hW1b   = (float*)alloc((size_t)B_ * H_ * 4);
  float* sc_u   = (float*)alloc((size_t)B_ * TU * 4);
  float* sc_b   = (float*)alloc((size_t)B_ * TB * 4);
  float* sc_p   = (float*)alloc((size_t)B_ * TP * 4);
  float* ctx_u  = (float*)alloc((size_t)B_ * H_ * 4);
  float* ctx_b  = (float*)alloc((size_t)B_ * H_ * 4);
  float* ctx_p  = (float*)alloc((size_t)B_ * H_ * 4);
  float* xbuf   = (float*)alloc((size_t)B_ * XDIM * 4);
  float* gi     = (float*)alloc((size_t)B_ * GH3 * 4);
  float* gh     = (float*)alloc((size_t)B_ * GH3 * 4);
  float* hnew   = (float*)alloc((size_t)B_ * H_ * 4);
  float* gen    = (float*)alloc((size_t)B_ * V_ * 4);
  float* cp_raw = (float*)alloc((size_t)B_ * TB * 4);

  auto packGrid = [](int ntiles, int kc) {
    long long t = (long long)ntiles * kc * 32;
    return (unsigned)((t + 255) / 256);
  };
  // 1) pack all weights into bf16 B-fragment order (L2-resident afterwards)
  pack_w_kernel<<<packGrid(32,  KC_H), 256, 0, stream>>>(attn_W,  2 * H_, 0,  H_,  32,  KC_H, W1p);
  pack_w_kernel<<<packGrid(32,  KC_H), 256, 0, stream>>>(attn_W,  2 * H_, H_, H_,  32,  KC_H, W2p);
  pack_w_kernel<<<packGrid(32,  KC_H), 256, 0, stream>>>(Wcopy_w, H_,     0,  H_,  32,  KC_H, Wcpyp);
  pack_w_kernel<<<packGrid(96,  KC_X), 256, 0, stream>>>(gru_W_ih, XDIM,  0,  GH3, 96,  KC_X, Wihp);
  pack_w_kernel<<<packGrid(96,  KC_H), 256, 0, stream>>>(gru_W_hh, H_,    0,  GH3, 96,  KC_H, Whhp);
  pack_w_kernel<<<packGrid(192, KC_H), 256, 0, stream>>>(Wgen_w,  H_,     0,  V_,  192, KC_H, Wgenp);

  // 2) hW1b = h0 @ W1ᵀ + attn_b  (shared across all attention targets)
  dim3 g1(B_ / 16, H_ / 128);
  gemm_bf16_kernel<<<g1, 256, 0, stream>>>(dec_last_h, H_, W1p, attn_b, hW1b, H_, B_, H_, H_);

  // 3) fused attention scores: sum_n tanh(encW2 + hW1b)·v_w  (no e tensor)
  fused_tanh_dot_kernel<<<B_ * TU / 16, 256, 0, stream>>>(usdx_h,   W2p, hW1b, H_, v_w, 0, sc_u, TU);
  fused_tanh_dot_kernel<<<B_ * TB / 16, 256, 0, stream>>>(bspn_h,   W2p, hW1b, H_, v_w, 0, sc_b, TB);
  fused_tanh_dot_kernel<<<B_ * TP / 16, 256, 0, stream>>>(pvaspn_h, W2p, hW1b, H_, v_w, 0, sc_p, TP);

  // 4) masked softmax + context vectors (async-to-LDS pipelined)
  softmax_ctx_kernel<<<B_, 256, 0, stream>>>(sc_u, usdx_ids,   usdx_h,   ctx_u, TU);
  softmax_ctx_kernel<<<B_, 256, 0, stream>>>(sc_b, bspn_ids,   bspn_h,   ctx_b, TB);
  softmax_ctx_kernel<<<B_, 256, 0, stream>>>(sc_p, pvaspn_ids, pvaspn_h, ctx_p, TP);

  // 5) GRU
  build_x_kernel<<<B_, 256, 0, stream>>>(dec_last_w, emb_table, ctx_u, ctx_b, ctx_p, db, xbuf);
  dim3 g2(B_ / 16, GH3 / 128);
  gemm_bf16_kernel<<<g2, 256, 0, stream>>>(xbuf,       XDIM, Wihp, gru_b_ih, gi, GH3, B_, GH3, XDIM);
  gemm_bf16_kernel<<<g2, 256, 0, stream>>>(dec_last_h, H_,   Whhp, gru_b_hh, gh, GH3, B_, GH3, H_);
  gru_gates_kernel<<<(B_ * H_) / 256, 256, 0, stream>>>(gi, gh, dec_last_h, hnew);

  // 6) generation logits + fused copy scores
  dim3 g3(B_ / 16, 24);
  gemm_bf16_kernel<<<g3, 256, 0, stream>>>(hnew, H_, Wgenp, Wgen_b, gen, V_, B_, V_, H_);
  fused_tanh_dot_kernel<<<B_ * TB / 16, 256, 0, stream>>>(bspn_h, Wcpyp, Wcopy_b, 0, hnew, H_, cp_raw, TB);

  // 7) log-softmax + OOV scatter + logaddexp merge
  finalize_kernel<<<B_, 256, 0, stream>>>(gen, cp_raw, bspn_ids, bspn_nounk, out);
}
